// ConvolutionalAttention2D_Old_43628277793388
// MI455X (gfx1250) — matrix operand order repaired and checked by reading, not compiled, not run on hardware
//
#include <hip/hip_runtime.h>
#include <hip/hip_bf16.h>

typedef __attribute__((ext_vector_type(16))) __bf16 v16bf;
typedef __attribute__((ext_vector_type(8)))  float  v8f;
typedef __attribute__((ext_vector_type(4)))  unsigned int v4u;

union Frag {
    v16bf bf;
    struct { v4u lo, hi; } u;
};

static constexpr int Cc = 256;     // channels
static constexpr int Bb = 16;      // batch
static constexpr int Nn = 4096;    // H*W
static constexpr int LDT = Cc + 8; // LDS transpose-tile stride (ushort units)
static constexpr int LDP = 40;     // LDS K-panel stride (ushort units): 32 + 8 pad

__device__ __forceinline__ unsigned short f2bf(float f) {
    unsigned int u = __float_as_uint(f);
    unsigned int r = (u + 0x7FFFu + ((u >> 16) & 1u)) >> 16;   // RNE
    return (unsigned short)r;
}

__device__ __forceinline__ float phi_fn(float x) {
    // elu(x)+1 : x>0 -> x+1 ; else exp(x)
    return x > 0.0f ? x + 1.0f : __expf(x);
}

// A fragment (16x32 bf16): row = lane%16; K chunks {0-7,16-23} (lanes<16) / {8-15,24-31}
__device__ __forceinline__ Frag loadA(const unsigned short* __restrict__ base,
                                      int ld, int m0, int k0) {
    int lane = threadIdx.x & 31;
    const unsigned short* p = base + (size_t)(m0 + (lane & 15)) * ld + k0 + ((lane >> 4) << 3);
    Frag f;
    f.u.lo = *(const v4u*)p;
    f.u.hi = *(const v4u*)(p + 16);
    return f;
}

// B fragment (32x16 bf16) from K-contiguous storage laid out [col][k] with stride ld:
// col = lane%16; K chunks {0-7,16-23} (lanes<16) / {8-15,24-31}
__device__ __forceinline__ Frag loadB(const unsigned short* __restrict__ base,
                                      int ld, int n0, int k0) {
    int lane = threadIdx.x & 31;
    const unsigned short* p = base + (size_t)(n0 + (lane & 15)) * ld + k0 + ((lane >> 4) << 3);
    Frag f;
    f.u.lo = *(const v4u*)p;
    f.u.hi = *(const v4u*)(p + 16);
    return f;
}

// Async DMA: 16 bytes global -> LDS per lane (tracked by ASYNCcnt, no VGPR return)
__device__ __forceinline__ void async_g2l_b128(unsigned lds_off, const void* gaddr) {
    asm volatile("global_load_async_to_lds_b128 %0, %1, off"
                 :: "v"(lds_off), "v"(gaddr) : "memory");
}
__device__ __forceinline__ void wait_async0() {
    asm volatile("s_wait_asynccnt 0" ::: "memory");
}

#define WMMA_BF16(A, B, Cacc) \
    __builtin_amdgcn_wmma_f32_16x16x32_bf16(false, (A).bf, false, (B).bf, (short)0, (Cacc), false, false)

// ---------------------------------------------------------------------------
// f32 -> bf16 weight conversion
__global__ void k_cvt_bf16(const float* __restrict__ s, unsigned short* __restrict__ d, int n) {
    int i = blockIdx.x * blockDim.x + threadIdx.x;
    if (i < n) d[i] = f2bf(s[i]);
}

// ---------------------------------------------------------------------------
// Fused q/k/v projection + phi, bf16 outputs.
// grid = (Nn/64, Bb), block = 256 (8 waves). Each wave owns 32 output rows.
__global__ __launch_bounds__(256) void k_qkv(
        const float* __restrict__ x,
        const unsigned short* __restrict__ Wq,
        const unsigned short* __restrict__ Wk,
        const unsigned short* __restrict__ Wv,
        unsigned short* __restrict__ phiQ,
        unsigned short* __restrict__ phiK,
        unsigned short* __restrict__ phiV) {
    __shared__ unsigned short xT[64 * LDT];   // x tile transposed: [n=64][c=256] bf16
    const int tid = threadIdx.x;
    const int n0  = blockIdx.x * 64;
    const int b   = blockIdx.y;
    const float* xb = x + (size_t)b * Cc * Nn + n0;

    // stage x[c, n0:n0+64] (f32) into LDS as bf16, transposed to K-contiguous
    {
        const int nch = (tid & 7) * 8;
        #pragma unroll
        for (int it = 0; it < 8; ++it) {
            int c = (tid >> 3) + it * 32;
            const float* p = xb + (size_t)c * Nn + nch;
            float4 f0 = *(const float4*)p;
            float4 f1 = *(const float4*)(p + 4);
            unsigned short* q = &xT[(size_t)nch * LDT + c];
            q[0 * LDT] = f2bf(f0.x); q[1 * LDT] = f2bf(f0.y);
            q[2 * LDT] = f2bf(f0.z); q[3 * LDT] = f2bf(f0.w);
            q[4 * LDT] = f2bf(f1.x); q[5 * LDT] = f2bf(f1.y);
            q[6 * LDT] = f2bf(f1.z); q[7 * LDT] = f2bf(f1.w);
        }
    }
    __syncthreads();

    const int wave = tid >> 5;
    const int lane = tid & 31;
    const int m0   = wave * 32;
    const unsigned short* Ws[3] = {Wq, Wk, Wv};
    unsigned short*       Os[3] = {phiQ, phiK, phiV};

    #pragma unroll
    for (int m = 0; m < 3; ++m) {
        const unsigned short* W = Ws[m];
        v8f zero = {0.f, 0.f, 0.f, 0.f, 0.f, 0.f, 0.f, 0.f};
        v8f acc[2][4];
        #pragma unroll
        for (int mi = 0; mi < 2; ++mi)
            #pragma unroll
            for (int j = 0; j < 4; ++j) acc[mi][j] = zero;

        #pragma unroll 2
        for (int k0 = 0; k0 < Cc; k0 += 32) {
            Frag a0 = loadA(W, Cc, m0, k0);
            Frag a1 = loadA(W, Cc, m0 + 16, k0);
            #pragma unroll
            for (int j = 0; j < 4; ++j) {
                Frag bb = loadB(xT, LDT, j * 16, k0);
                acc[0][j] = WMMA_BF16(a0, bb, acc[0][j]);
                acc[1][j] = WMMA_BF16(a1, bb, acc[1][j]);
            }
        }

        unsigned short* out = Os[m] + (size_t)b * Cc * Nn;
        const int rowadd = (lane >> 4) << 3;
        const int ncol   = n0 + (lane & 15);
        #pragma unroll
        for (int mi = 0; mi < 2; ++mi)
            #pragma unroll
            for (int j = 0; j < 4; ++j)
                #pragma unroll
                for (int r = 0; r < 8; ++r) {
                    int o = m0 + mi * 16 + rowadd + r;
                    out[(size_t)o * Nn + ncol + j * 16] = f2bf(phi_fn(acc[mi][j][r]));
                }
    }
}

// ---------------------------------------------------------------------------
// qv[b] = phiQ[b] (CxN) @ phiV[b]^T (NxC), f32 out.
// grid = (2, 2, Bb) of 128x128 tiles, block 256 = 8 waves (4x2), wave = 32x64.
// K-panels of phiQ/phiV are DMA'd to LDS with async global->LDS copies
// (ASYNCcnt path); both operands are K-contiguous in memory so no transpose
// is needed, and all 8 waves share one 16 KB panel instead of 48 KB of
// per-wave L2 reads per K-step.
__global__ __launch_bounds__(256) void k_qv(
        const unsigned short* __restrict__ phiQ,
        const unsigned short* __restrict__ phiV,
        float* __restrict__ qv) {
    __shared__ unsigned short Ash[128 * LDP];   // phiQ panel: [row 0..127][k 0..31]
    __shared__ unsigned short Bsh[128 * LDP];   // phiV panel: [col 0..127][k 0..31]
    const int b    = blockIdx.z;
    const int row0 = blockIdx.y * 128;
    const int col0 = blockIdx.x * 128;
    const int tid  = threadIdx.x;
    const int wave = tid >> 5;
    const int lane = tid & 31;
    const int m0l  = (wave & 3) * 32;   // local row base within panel
    const int c0l  = (wave >> 2) * 64;  // local col base within panel
    const unsigned short* Q = phiQ + (size_t)b * Cc * Nn;
    const unsigned short* V = phiV + (size_t)b * Cc * Nn;

    // staging chunk coords: 512 chunks of 16B per panel; this thread owns 2
    const int r0c = (tid)        >> 2, k0c = ((tid)        & 3) * 8;
    const int r1c = (tid + 256)  >> 2, k1c = ((tid + 256)  & 3) * 8;
    const unsigned a0off = (unsigned)(uintptr_t)&Ash[r0c * LDP + k0c];
    const unsigned a1off = (unsigned)(uintptr_t)&Ash[r1c * LDP + k1c];
    const unsigned b0off = (unsigned)(uintptr_t)&Bsh[r0c * LDP + k0c];
    const unsigned b1off = (unsigned)(uintptr_t)&Bsh[r1c * LDP + k1c];

    v8f zero = {0.f, 0.f, 0.f, 0.f, 0.f, 0.f, 0.f, 0.f};
    v8f acc[2][4];
    #pragma unroll
    for (int mi = 0; mi < 2; ++mi)
        #pragma unroll
        for (int j = 0; j < 4; ++j) acc[mi][j] = zero;

    for (int k0 = 0; k0 < Nn; k0 += 32) {
        __syncthreads();   // all waves done reading previous panel
        async_g2l_b128(a0off, Q + (size_t)(row0 + r0c) * Nn + k0 + k0c);
        async_g2l_b128(a1off, Q + (size_t)(row0 + r1c) * Nn + k0 + k1c);
        async_g2l_b128(b0off, V + (size_t)(col0 + r0c) * Nn + k0 + k0c);
        async_g2l_b128(b1off, V + (size_t)(col0 + r1c) * Nn + k0 + k1c);
        if (k0 + 512 < Nn) {  // warm L2 ahead of the DMA engine: global_prefetch_b8
            __builtin_prefetch((const void*)(Q + (size_t)(row0 + r0c) * Nn + k0 + 512), 0, 3);
            __builtin_prefetch((const void*)(V + (size_t)(col0 + r0c) * Nn + k0 + 512), 0, 3);
        }
        wait_async0();     // this wave's DMA done
        __syncthreads();   // panel visible to all waves

        Frag a0 = loadA(Ash, LDP, m0l, 0);
        Frag a1 = loadA(Ash, LDP, m0l + 16, 0);
        #pragma unroll
        for (int j = 0; j < 4; ++j) {
            Frag bb = loadB(Bsh, LDP, c0l + j * 16, 0);
            acc[0][j] = WMMA_BF16(a0, bb, acc[0][j]);
            acc[1][j] = WMMA_BF16(a1, bb, acc[1][j]);
        }
    }

    float* out = qv + (size_t)b * Cc * Cc;
    const int rowadd = (lane >> 4) << 3;
    const int cc     = lane & 15;
    #pragma unroll
    for (int mi = 0; mi < 2; ++mi)
        #pragma unroll
        for (int j = 0; j < 4; ++j)
            #pragma unroll
            for (int r = 0; r < 8; ++r) {
                int o = row0 + m0l + mi * 16 + rowadd + r;
                out[(size_t)o * Cc + col0 + c0l + j * 16 + cc] = acc[mi][j][r];
            }
}

// ---------------------------------------------------------------------------
// M[b] = Wo @ qv[b]  (f32 math, tiny: 0.5 GFLOP total), stored bf16.
__global__ __launch_bounds__(256) void k_m(
        const float* __restrict__ Wo,
        const float* __restrict__ qv,
        unsigned short* __restrict__ M) {
    int idx = blockIdx.x * blockDim.x + threadIdx.x;   // [b][o][d]
    int b = idx >> 16;
    int o = (idx >> 8) & 255;
    int d = idx & 255;
    const float* q = qv + (size_t)b * Cc * Cc + d;
    const float* w = Wo + (size_t)o * Cc;
    float acc = 0.0f;
    #pragma unroll 8
    for (int c = 0; c < Cc; ++c) acc = fmaf(w[c], q[(size_t)c * Cc], acc);
    M[idx] = f2bf(acc);
}

// ---------------------------------------------------------------------------
// out[b] = M[b] @ phiK[b] + bo   (fused Wo@(qv@phiK)), f32 out.
// grid = (Nn/64, Bb), block 256.
__global__ __launch_bounds__(256) void k_out(
        const unsigned short* __restrict__ Mall,
        const unsigned short* __restrict__ phiK,
        const float* __restrict__ bo,
        float* __restrict__ out) {
    __shared__ unsigned short kT[64 * LDT];   // phiK tile transposed: [n=64][d=256] bf16
    const int tid = threadIdx.x;
    const int n0  = blockIdx.x * 64;
    const int b   = blockIdx.y;
    const unsigned short* Kb = phiK + (size_t)b * Cc * Nn + n0;

    {
        const int nch = (tid & 7) * 8;
        #pragma unroll
        for (int it = 0; it < 8; ++it) {
            int d = (tid >> 3) + it * 32;
            v4u v = *(const v4u*)(Kb + (size_t)d * Nn + nch);
            const unsigned short* s = (const unsigned short*)&v;
            unsigned short* q = &kT[(size_t)nch * LDT + d];
            #pragma unroll
            for (int i = 0; i < 8; ++i) q[i * LDT] = s[i];
        }
    }
    __syncthreads();

    const int wave = tid >> 5;
    const int lane = tid & 31;
    const int m0   = wave * 32;
    const unsigned short* A = Mall + (size_t)b * Cc * Cc;

    v8f zero = {0.f, 0.f, 0.f, 0.f, 0.f, 0.f, 0.f, 0.f};
    v8f acc[2][4];
    #pragma unroll
    for (int mi = 0; mi < 2; ++mi)
        #pragma unroll
        for (int j = 0; j < 4; ++j) acc[mi][j] = zero;

    #pragma unroll 2
    for (int k0 = 0; k0 < Cc; k0 += 32) {
        Frag a0 = loadA(A, Cc, m0, k0);
        Frag a1 = loadA(A, Cc, m0 + 16, k0);
        #pragma unroll
        for (int j = 0; j < 4; ++j) {
            Frag bb = loadB(kT, LDT, j * 16, k0);
            acc[0][j] = WMMA_BF16(a0, bb, acc[0][j]);
            acc[1][j] = WMMA_BF16(a1, bb, acc[1][j]);
        }
    }

    float* ob = out + (size_t)b * Cc * Nn;
    const int rowadd = (lane >> 4) << 3;
    const int nc     = n0 + (lane & 15);
    #pragma unroll
    for (int mi = 0; mi < 2; ++mi)
        #pragma unroll
        for (int j = 0; j < 4; ++j)
            #pragma unroll
            for (int r = 0; r < 8; ++r) {
                int o = m0 + mi * 16 + rowadd + r;
                ob[(size_t)o * Nn + nc + j * 16] = acc[mi][j][r] + bo[o];
            }
}

// ---------------------------------------------------------------------------
extern "C" void kernel_launch(void* const* d_in, const int* in_sizes, int n_in,
                              void* d_out, int out_size, void* d_ws, size_t ws_size,
                              hipStream_t stream) {
    const float* x  = (const float*)d_in[0];
    const float* Wq = (const float*)d_in[1];
    const float* Wk = (const float*)d_in[2];
    const float* Wv = (const float*)d_in[3];
    const float* Wo = (const float*)d_in[4];
    const float* bo = (const float*)d_in[5];
    float* out = (float*)d_out;

    char* ws = (char*)d_ws;
    const size_t SZphi = (size_t)Bb * Cc * Nn * sizeof(unsigned short);   // 32 MB
    unsigned short* phiQ = (unsigned short*)(ws);
    unsigned short* phiK = (unsigned short*)(ws + SZphi);
    unsigned short* phiV = (unsigned short*)(ws + 2 * SZphi);
    float*          qv   = (float*)(ws + 3 * SZphi);                       // 4 MB
    unsigned short* Mb   = (unsigned short*)(ws + 3 * SZphi + (size_t)Bb * Cc * Cc * 4);
    unsigned short* wqb  = (unsigned short*)(ws + 3 * SZphi + (size_t)Bb * Cc * Cc * 6);
    unsigned short* wkb  = wqb + Cc * Cc;
    unsigned short* wvb  = wkb + Cc * Cc;

    const int wn = Cc * Cc;
    k_cvt_bf16<<<(wn + 255) / 256, 256, 0, stream>>>(Wq, wqb, wn);
    k_cvt_bf16<<<(wn + 255) / 256, 256, 0, stream>>>(Wk, wkb, wn);
    k_cvt_bf16<<<(wn + 255) / 256, 256, 0, stream>>>(Wv, wvb, wn);

    k_qkv<<<dim3(Nn / 64, Bb), 256, 0, stream>>>(x, wqb, wkb, wvb, phiQ, phiK, phiV);
    k_qv <<<dim3(2, 2, Bb),   256, 0, stream>>>(phiQ, phiV, qv);
    k_m  <<<(Bb * Cc * Cc) / 256, 256, 0, stream>>>(Wo, qv, Mb);
    k_out<<<dim3(Nn / 64, Bb), 256, 0, stream>>>(Mb, phiK, bo, out);
}
